// ramsey_MPNN_3942779978324
// MI455X (gfx1250) — matrix operand order repaired and checked by reading, not compiled, not run on hardware
//
#include <hip/hip_runtime.h>
#include <hip/hip_bf16.h>

#define NN 1536
#define FF 32
#define HH 64
#define NEDGE ((NN * (NN - 1)) / 2)   // 1178880
#define NTILES2 (NEDGE / 32)          // 36840 dual-tiles (32 edges each)

typedef __attribute__((ext_vector_type(16))) _Float16 v16h;
typedef __attribute__((ext_vector_type(8)))  _Float16 h8;
typedef __attribute__((ext_vector_type(8)))  float    v8f;

__device__ __forceinline__ float lrelu_f(float x) { return x >= 0.f ? x : 0.01f * x; }
__device__ __forceinline__ float relu_f(float x)  { return x > 0.f ? x : 0.f; }

// ---------------------------------------------------------------------------
// misc small kernels
// ---------------------------------------------------------------------------
__global__ void fill_zero(float* __restrict__ p, int n) {
  for (int i = blockIdx.x * blockDim.x + threadIdx.x; i < n; i += gridDim.x * blockDim.x)
    p[i] = 0.f;
}

// W [K][Nc] f32 row-major  ->  Wt [Nc][K] f16 row-major (K contiguous per out-channel)
__global__ void transpose_f16(const float* __restrict__ W, _Float16* __restrict__ Wt,
                              int K, int Nc) {
  int idx = blockIdx.x * blockDim.x + threadIdx.x;
  if (idx < K * Nc) {
    int n = idx / K, k = idx % K;
    Wt[idx] = (_Float16)W[k * Nc + n];
  }
}

// out[r] = (1+eps)*x[r] + sum_{s<r} x[s]   (per feature column; block per column)
__global__ __launch_bounds__(256)
void prefix_gin(const float* __restrict__ X, const float* __restrict__ epsp,
                float* __restrict__ Y, int C) {
  int c = blockIdx.x, t = threadIdx.x;
  __shared__ float sc[256];
  float v[6];
  float s = 0.f;
#pragma unroll
  for (int q = 0; q < 6; q++) { v[q] = X[(t * 6 + q) * C + c]; s += v[q]; }
  sc[t] = s;
  __syncthreads();
  for (int off = 1; off < 256; off <<= 1) {
    float add = (t >= off) ? sc[t - off] : 0.f;
    __syncthreads();
    sc[t] += add;
    __syncthreads();
  }
  float run = sc[t] - s;  // exclusive prefix over earlier threads
  float e1 = 1.f + epsp[0];
#pragma unroll
  for (int q = 0; q < 6; q++) {
    Y[(t * 6 + q) * C + c] = e1 * v[q] + run;
    run += v[q];
  }
}

// BatchNorm (training stats, biased var) + leaky relu; block per channel
__global__ __launch_bounds__(256)
void bn_lrelu(const float* __restrict__ X, const float* __restrict__ g,
              const float* __restrict__ be, float* __restrict__ Y) {
  int c = blockIdx.x, t = threadIdx.x;
  __shared__ float s1[256], s2[256];
  float v[6];
  float a = 0.f, b = 0.f;
#pragma unroll
  for (int q = 0; q < 6; q++) {
    float x = X[(t * 6 + q) * HH + c];
    v[q] = x; a += x; b += x * x;
  }
  s1[t] = a; s2[t] = b;
  __syncthreads();
  for (int off = 128; off > 0; off >>= 1) {
    if (t < off) { s1[t] += s1[t + off]; s2[t] += s2[t + off]; }
    __syncthreads();
  }
  float mean = s1[0] * (1.f / NN);
  float var  = s2[0] * (1.f / NN) - mean * mean;
  float inv  = rsqrtf(var + 1e-5f);
  float gg = g[c], bb = be[c];
#pragma unroll
  for (int q = 0; q < 6; q++)
    Y[(t * 6 + q) * HH + c] = lrelu_f((v[q] - mean) * inv * gg + bb);
}

// ---------------------------------------------------------------------------
// Node GEMM: C[N x Cout] = act(A[N x K] @ W + bias) (+resid), WMMA f16->f32.
// One wave per 16x16 output tile. Wt is f16 [Cout][K] (K contiguous).
// ---------------------------------------------------------------------------
__global__ __launch_bounds__(32)
void gemm_node(const float* __restrict__ A, int K,
               const _Float16* __restrict__ Wt, const float* __restrict__ bias,
               float* __restrict__ C, int Cout, int actMode,
               const float* __restrict__ resid, _Float16* __restrict__ C16) {
  int tm = blockIdx.x, tn = blockIdx.y;
  int l = threadIdx.x, m = l & 15, hi = l >> 4;
  v8f acc = {};
  for (int kb = 0; kb < K; kb += 32) {
    v16h a;
    const float* ap = A + (tm * 16 + m) * K + kb + hi * 8;
#pragma unroll
    for (int q = 0; q < 8; q++) a[q] = (_Float16)ap[q];
#pragma unroll
    for (int q = 0; q < 8; q++) a[8 + q] = (_Float16)ap[16 + q];
    v16h b;
    const _Float16* bp = Wt + (tn * 16 + m) * K + kb + hi * 8;
    h8 b0 = *(const h8*)bp;
    h8 b1 = *(const h8*)(bp + 16);
#pragma unroll
    for (int q = 0; q < 8; q++) { b[q] = b0[q]; b[8 + q] = b1[q]; }
    acc = __builtin_amdgcn_wmma_f32_16x16x32_f16(false, a, false, b, (short)0, acc,
                                                 false, false);
  }
  int col = tn * 16 + m;
  float bv = bias[col];
#pragma unroll
  for (int j = 0; j < 8; j++) {
    int row = tm * 16 + j + hi * 8;
    float v = acc[j] + bv;
    if (actMode == 0) v = relu_f(v);
    else if (actMode == 1) v = lrelu_f(v);
    if (resid) v += resid[row * Cout + col];
    C[row * Cout + col] = v;
    if (C16) C16[row * Cout + col] = (_Float16)v;
  }
}

// ---------------------------------------------------------------------------
// Edge predictor. One wave = 32 edges (two 16-edge sub-tiles) per iteration.
// Layers computed as Y^T = W^T x X^T: A = weight tile (LDS, reused for BOTH
// sub-tiles -> 1 DS fragment load per WMMA), B = activations (N = edge).
// D layout (lane = edge) feeds the next layer's B fragment via shfl_xor(16).
// ---------------------------------------------------------------------------
__device__ __forceinline__ v16h load_wfrag(const _Float16* __restrict__ W,
                                           int t, int kchunk, int m, int hi) {
  const _Float16* p = W + (t * 16 + m) * HH + kchunk * 32 + hi * 16;
  h8 u0 = *(const h8*)p;
  h8 u1 = *(const h8*)(p + 8);
  v16h a;
#pragma unroll
  for (int q = 0; q < 8; q++) { a[q] = u0[q]; a[8 + q] = u1[q]; }
  return a;
}

__device__ __forceinline__ v16h load_xfrag(const _Float16* __restrict__ h16,
                                           int node, int hi) {
  const _Float16* p = h16 + node * FF + hi * 16;
  h8 u0 = *(const h8*)p;
  h8 u1 = *(const h8*)(p + 8);
  v16h b;
#pragma unroll
  for (int q = 0; q < 8; q++) { b[q] = u0[q]; b[8 + q] = u1[q]; }
  return b;
}

__device__ __forceinline__ v16h build_bfrag(const float af[4][8], int c, int hi) {
  // chunk c halves h: K = 32c + h + hi*16 (feature index of previous layer)
  v16h b;
#pragma unroll
  for (int h = 0; h < 8; h++) {
    float p0 = __shfl_xor(af[2 * c + 1][h], 16, 32);
    float p1 = __shfl_xor(af[2 * c][h], 16, 32);
    float lo = hi ? p0 : af[2 * c][h];
    float hv = hi ? af[2 * c + 1][h] : p1;
    b[h] = (_Float16)lo;
    b[8 + h] = (_Float16)hv;
  }
  return b;
}

__device__ __forceinline__ void decode_edge(int e, int& io, int& jo) {
  int rad = (2 * NN - 1) * (2 * NN - 1) - 8 * e;   // exact in fp32 range (< 2^24)
  float sf = sqrtf((float)rad);
  int i = (int)(((float)(2 * NN - 1) - sf) * 0.5f);
  if (i < 0) i = 0;
  if (i > NN - 2) i = NN - 2;
  while (i < NN - 2 && (((i + 1) * (2 * NN - 1 - (i + 1))) >> 1) <= e) ++i;
  while (i > 0 && ((i * (2 * NN - 1 - i)) >> 1) > e) --i;
  int base = (i * (2 * NN - 1 - i)) >> 1;
  io = i;
  jo = i + 1 + (e - base);
}

__global__ __launch_bounds__(256)
void edge_mlp(const _Float16* __restrict__ h16,
              const _Float16* __restrict__ w1t, const _Float16* __restrict__ w2t,
              const _Float16* __restrict__ w3t,
              const float* __restrict__ b1, const float* __restrict__ b2,
              const float* __restrict__ b3,
              const float* __restrict__ w4, const float* __restrict__ b4,
              float* __restrict__ probs) {
  __shared__ _Float16 sW[3][HH * HH];
  __shared__ float sB[3][HH];
  __shared__ float sW4[HH];
  __shared__ float sB4;
  for (int idx = threadIdx.x; idx < HH * HH; idx += blockDim.x) {
    sW[0][idx] = w1t[idx];
    sW[1][idx] = w2t[idx];
    sW[2][idx] = w3t[idx];
  }
  if (threadIdx.x < HH) {
    sB[0][threadIdx.x] = b1[threadIdx.x];
    sB[1][threadIdx.x] = b2[threadIdx.x];
    sB[2][threadIdx.x] = b3[threadIdx.x];
    sW4[threadIdx.x] = w4[threadIdx.x];
  }
  if (threadIdx.x == 0) sB4 = b4[0];
  __syncthreads();

  int lane = threadIdx.x & 31;
  int m = lane & 15, hi = lane >> 4;
  int wid = blockIdx.x * (blockDim.x >> 5) + (threadIdx.x >> 5);
  int nw = gridDim.x * (blockDim.x >> 5);

  for (int tile = wid; tile < NTILES2; tile += nw) {
    int eA = tile * 32 + m;
    int eB = eA + 16;
    int iA, jA, iB, jB;
    decode_edge(eA, iA, jA);
    decode_edge(eB, iB, jB);

    // layer-1 B fragments: ef = [h16[i] (K 0..31) | h16[j] (K 32..63)]
    v16h bxA0 = load_xfrag(h16, iA, hi);
    v16h bxA1 = load_xfrag(h16, jA, hi);
    v16h bxB0 = load_xfrag(h16, iB, hi);
    v16h bxB1 = load_xfrag(h16, jB, hi);

    float afA[4][8], afB[4][8];
    // ---- layer 1 ----
#pragma unroll
    for (int t = 0; t < 4; t++) {
      v16h w0 = load_wfrag(sW[0], t, 0, m, hi);
      v16h w1 = load_wfrag(sW[0], t, 1, m, hi);
      v8f a0 = {}, a1 = {};
      a0 = __builtin_amdgcn_wmma_f32_16x16x32_f16(false, w0, false, bxA0, (short)0, a0, false, false);
      a1 = __builtin_amdgcn_wmma_f32_16x16x32_f16(false, w0, false, bxB0, (short)0, a1, false, false);
      a0 = __builtin_amdgcn_wmma_f32_16x16x32_f16(false, w1, false, bxA1, (short)0, a0, false, false);
      a1 = __builtin_amdgcn_wmma_f32_16x16x32_f16(false, w1, false, bxB1, (short)0, a1, false, false);
#pragma unroll
      for (int q = 0; q < 8; q++) {
        float bb = sB[0][16 * t + q + hi * 8];
        afA[t][q] = relu_f(a0[q] + bb);
        afB[t][q] = relu_f(a1[q] + bb);
      }
    }
    // ---- layers 2, 3 ----
    for (int L = 1; L < 3; L++) {
      v16h nbA0 = build_bfrag(afA, 0, hi);
      v16h nbA1 = build_bfrag(afA, 1, hi);
      v16h nbB0 = build_bfrag(afB, 0, hi);
      v16h nbB1 = build_bfrag(afB, 1, hi);
      float nfA[4][8], nfB[4][8];
#pragma unroll
      for (int t = 0; t < 4; t++) {
        v16h w0 = load_wfrag(sW[L], t, 0, m, hi);
        v16h w1 = load_wfrag(sW[L], t, 1, m, hi);
        v8f a0 = {}, a1 = {};
        a0 = __builtin_amdgcn_wmma_f32_16x16x32_f16(false, w0, false, nbA0, (short)0, a0, false, false);
        a1 = __builtin_amdgcn_wmma_f32_16x16x32_f16(false, w0, false, nbB0, (short)0, a1, false, false);
        a0 = __builtin_amdgcn_wmma_f32_16x16x32_f16(false, w1, false, nbA1, (short)0, a0, false, false);
        a1 = __builtin_amdgcn_wmma_f32_16x16x32_f16(false, w1, false, nbB1, (short)0, a1, false, false);
#pragma unroll
        for (int q = 0; q < 8; q++) {
          float bb = sB[L][16 * t + q + hi * 8];
          nfA[t][q] = relu_f(a0[q] + bb);
          nfB[t][q] = relu_f(a1[q] + bb);
        }
      }
#pragma unroll
      for (int t = 0; t < 4; t++)
#pragma unroll
        for (int q = 0; q < 8; q++) { afA[t][q] = nfA[t][q]; afB[t][q] = nfB[t][q]; }
    }
    // ---- final 64 -> 1 dot, sigmoid, symmetric scatter ----
    float sA = 0.f, sBv = 0.f;
#pragma unroll
    for (int t = 0; t < 4; t++)
#pragma unroll
      for (int q = 0; q < 8; q++) {
        float wv = sW4[16 * t + q + hi * 8];
        sA  += afA[t][q] * wv;
        sBv += afB[t][q] * wv;
      }
    sA  += __shfl_xor(sA, 16, 32);
    sBv += __shfl_xor(sBv, 16, 32);
    if (hi == 0) {
      float pA = 1.f / (1.f + __expf(-(sA + sB4)));
      float pB = 1.f / (1.f + __expf(-(sBv + sB4)));
      probs[iA * NN + jA] = pA;
      probs[jA * NN + iA] = pA;
      probs[iB * NN + jB] = pB;
      probs[jB * NN + iB] = pB;
    }
  }
}

// ---------------------------------------------------------------------------
extern "C" void kernel_launch(void* const* d_in, const int* in_sizes, int n_in,
                              void* d_out, int out_size, void* d_ws, size_t ws_size,
                              hipStream_t stream) {
  (void)in_sizes; (void)n_in; (void)out_size; (void)ws_size;
  const float* nodef  = (const float*)d_in[1];
  const float* c1_eps = (const float*)d_in[2];
  const float* c1_w1  = (const float*)d_in[3];
  const float* c1_b1  = (const float*)d_in[4];
  const float* c1_w2  = (const float*)d_in[5];
  const float* c1_b2  = (const float*)d_in[6];
  const float* c1_g   = (const float*)d_in[7];
  const float* c1_be  = (const float*)d_in[8];
  const float* cv_eps = (const float*)d_in[9];
  const float* cv_w1  = (const float*)d_in[10];
  const float* cv_b1  = (const float*)d_in[11];
  const float* cv_w2  = (const float*)d_in[12];
  const float* cv_b2  = (const float*)d_in[13];
  const float* cv_g   = (const float*)d_in[14];
  const float* cv_be  = (const float*)d_in[15];
  const float* lin1_w = (const float*)d_in[16];
  const float* lin1_b = (const float*)d_in[17];
  const float* lin2_w = (const float*)d_in[18];
  const float* lin2_b = (const float*)d_in[19];
  const float* ep_w1  = (const float*)d_in[20];
  const float* ep_b1  = (const float*)d_in[21];
  const float* ep_w2  = (const float*)d_in[22];
  const float* ep_b2  = (const float*)d_in[23];
  const float* ep_w3  = (const float*)d_in[24];
  const float* ep_b3  = (const float*)d_in[25];
  const float* ep_w4  = (const float*)d_in[26];
  const float* ep_b4  = (const float*)d_in[27];
  float* probs = (float*)d_out;

  char* ws = (char*)d_ws;
  float* t0   = (float*)(ws);
  float* t1   = (float*)(ws + 393216);
  float* t2   = (float*)(ws + 786432);
  float* hcur = (float*)(ws + 1179648);
  _Float16* h16 = (_Float16*)(ws + 1572864);
  _Float16* wf  = (_Float16*)(ws + 1671168);
  _Float16* c1w1t  = wf;
  _Float16* c1w2t  = c1w1t + 64 * 32;
  _Float16* cvw1t0 = c1w2t + 64 * 64;
  _Float16* cvw2t0 = cvw1t0 + 64 * 64;
  _Float16* cvw1t1 = cvw2t0 + 64 * 64;
  _Float16* cvw2t1 = cvw1t1 + 64 * 64;
  _Float16* lin1t  = cvw2t1 + 64 * 64;
  _Float16* lin2t  = lin1t + 64 * 64;
  _Float16* ep1t   = lin2t + 32 * 64;
  _Float16* ep2t   = ep1t + 64 * 64;
  _Float16* ep3t   = ep2t + 64 * 64;

  fill_zero<<<2304, 256, 0, stream>>>(probs, NN * NN);

  transpose_f16<<<8, 256, 0, stream>>>(c1_w1, c1w1t, 32, 64);
  transpose_f16<<<16, 256, 0, stream>>>(c1_w2, c1w2t, 64, 64);
  transpose_f16<<<16, 256, 0, stream>>>(cv_w1, cvw1t0, 64, 64);
  transpose_f16<<<16, 256, 0, stream>>>(cv_w1 + 4096, cvw1t1, 64, 64);
  transpose_f16<<<16, 256, 0, stream>>>(cv_w2, cvw2t0, 64, 64);
  transpose_f16<<<16, 256, 0, stream>>>(cv_w2 + 4096, cvw2t1, 64, 64);
  transpose_f16<<<16, 256, 0, stream>>>(lin1_w, lin1t, 64, 64);
  transpose_f16<<<8, 256, 0, stream>>>(lin2_w, lin2t, 64, 32);
  transpose_f16<<<16, 256, 0, stream>>>(ep_w1, ep1t, 64, 64);
  transpose_f16<<<16, 256, 0, stream>>>(ep_w2, ep2t, 64, 64);
  transpose_f16<<<16, 256, 0, stream>>>(ep_w3, ep3t, 64, 64);

  // GIN layer 1 (F -> H)
  prefix_gin<<<32, 256, 0, stream>>>(nodef, c1_eps, t0, 32);
  gemm_node<<<dim3(96, 4), 32, 0, stream>>>(t0, 32, c1w1t, c1_b1, t1, 64, 0, nullptr, nullptr);
  gemm_node<<<dim3(96, 4), 32, 0, stream>>>(t1, 64, c1w2t, c1_b2, t2, 64, 0, nullptr, nullptr);
  bn_lrelu<<<64, 256, 0, stream>>>(t2, c1_g, c1_be, hcur);
  // GIN layers 2, 3 (H -> H)
  for (int i = 0; i < 2; i++) {
    const _Float16* w1t = i ? cvw1t1 : cvw1t0;
    const _Float16* w2t = i ? cvw2t1 : cvw2t0;
    prefix_gin<<<64, 256, 0, stream>>>(hcur, cv_eps + i, t0, 64);
    gemm_node<<<dim3(96, 4), 32, 0, stream>>>(t0, 64, w1t, cv_b1 + 64 * i, t1, 64, 0, nullptr, nullptr);
    gemm_node<<<dim3(96, 4), 32, 0, stream>>>(t1, 64, w2t, cv_b2 + 64 * i, t2, 64, 0, nullptr, nullptr);
    bn_lrelu<<<64, 256, 0, stream>>>(t2, cv_g + 64 * i, cv_be + 64 * i, hcur);
  }
  // projection head: lrelu(lin1), lrelu(lin2) + residual, emit f16 table
  gemm_node<<<dim3(96, 4), 32, 0, stream>>>(hcur, 64, lin1t, lin1_b, t1, 64, 1, nullptr, nullptr);
  gemm_node<<<dim3(96, 2), 32, 0, stream>>>(t1, 64, lin2t, lin2_b, t2, 32, 1, nodef, h16);
  // edge predictor over all 1,178,880 edges (36,840 dual-tiles of 32 edges)
  edge_mlp<<<1152, 256, 0, stream>>>(h16, ep1t, ep2t, ep3t, ep_b1, ep_b2, ep_b3,
                                     ep_w4, ep_b4, probs);
}
